// Mamba_8143257993724
// MI455X (gfx1250) — matrix operand order repaired and checked by reading, not compiled, not run on hardware
//
#include <hip/hip_runtime.h>
#include <hip/hip_bf16.h>
#include <math.h>

// MI455X (gfx1250) implementation.
//   Phase 1 (precompute_kernel): conv1d(K=4)+SiLU, then the 4 head-wise
//     32x32 projections as f16 WMMA (v_wmma_f32_16x16x32_f16; DH=32 == one
//     K-step). Writes pre-activations (bias folded) as float4[g] in
//     [s][b][n][i] order -> one b128 load per lane per scan step.
//   Phase 2 (scan_kernel): one wave32 per (batch,head) recurrence. R kept in
//     VGPRs (128 f32/lane), h[j] broadcast via v_readlane (SGPR FMA operand),
//     8-deep register prefetch ring over the pre stream, split even/odd
//     accumulators for 8-way FMA chain ILP, GroupNorm over DH=32 fused via
//     wave shfl_xor reductions.

typedef __attribute__((ext_vector_type(16))) _Float16 v16h;
typedef __attribute__((ext_vector_type(8)))  float    v8f;

#define B_   8
#define D_   256
#define S_   4096      // 64*64
#define NH_  8
#define DH_  32
#define TS   32        // sequence rows per block in kernel 1
#define SROW 264       // padded LDS row stride in halfs (256 + 8 -> no bank conflicts)
#define PF   8         // prefetch depth in the scan

// ---- WMMA fragment helpers -------------------------------------------------
// A-matrix (16x32 f16) lane layout per CDNA5 ISA 7.12.2:
//   lane holds row M = lane&15; VGPR p<4: K = 2p,2p+1 (+8 if lane>=16),
//   VGPR p>=4: K = 16+2(p-4),+1 (+8 if lane>=16).
__device__ inline v16h loadA_lds(const _Float16* sh, int row, int lhalf) {
  v16h a;
#pragma unroll
  for (int p = 0; p < 8; ++p) {
    const int k = 2 * p + ((p >= 4) ? 8 : 0) + 8 * lhalf;
    a[2 * p]     = sh[row * SROW + k];
    a[2 * p + 1] = sh[row * SROW + k + 1];
  }
  return a;
}

// B-matrix (32x16 f16): mirror of the documented 64x16 sparse-B pattern:
//   lane holds column N = lane&15; VGPR p<4: K = 2p,2p+1 (+8 if lane>=16),
//   VGPR p>=4: K = 16+2(p-4),+1 (+8 if lane>=16).
// Needed value B[k][o] = W[o][k] (out[m][o] = sum_k A[m][k]*W[o][k]).
__device__ inline v16h loadB_w(const float* __restrict__ W, int o, int lhalf) {
  const float* r = W + (size_t)o * DH_;
  v16h bf;
#pragma unroll
  for (int p = 0; p < 8; ++p) {
    const int k = 2 * p + ((p >= 4) ? 8 : 0) + 8 * lhalf;
    const float2 v = *(const float2*)(r + k);   // k even -> 8B aligned
    bf[2 * p]     = (_Float16)v.x;
    bf[2 * p + 1] = (_Float16)v.y;
  }
  return bf;
}

__device__ inline v8f wmma_f16(v16h a, v16h b, v8f c) {
  return __builtin_amdgcn_wmma_f32_16x16x32_f16(false, a, false, b,
                                                (short)0, c, false, false);
}

// ---------------- Kernel 1: conv + SiLU + head-wise GEMMs -------------------
__global__ __launch_bounds__(256)
void precompute_kernel(const float* __restrict__ x,
                       const float* __restrict__ conv_w,
                       const float* __restrict__ conv_b,
                       const float* __restrict__ w_i,
                       const float* __restrict__ w_f,
                       const float* __restrict__ w_z,
                       const float* __restrict__ w_o,
                       const float* __restrict__ bias,
                       float4* __restrict__ pre4)
{
  __shared__ _Float16 sCH[TS * SROW];   // conv+SiLU rows (f16)
  __shared__ _Float16 sXH[TS * SROW];   // raw x rows (f16)

  const int blk = blockIdx.x;
  const int b   = blk >> 7;             // 128 s-blocks of 32 per batch
  const int s0  = (blk & 127) * TS;

  // Stage 1: one channel per thread, sliding-window causal conv over 32 rows.
  {
    const int d = threadIdx.x;                       // 0..255
    const float4 cw = ((const float4*)conv_w)[d];    // conv_w[d][0..3]
    const float  cb = conv_b[d];
    const float* xb = x + ((size_t)b * S_) * D_ + d;
    float w0 = (s0 - 3 >= 0) ? xb[(size_t)(s0 - 3) * D_] : 0.f;
    float w1 = (s0 - 2 >= 0) ? xb[(size_t)(s0 - 2) * D_] : 0.f;
    float w2 = (s0 - 1 >= 0) ? xb[(size_t)(s0 - 1) * D_] : 0.f;
    float w3 = xb[(size_t)s0 * D_];
#pragma unroll 4
    for (int si = 0; si < TS; ++si) {
      float xc = cw.x * w0 + cw.y * w1 + cw.z * w2 + cw.w * w3 + cb;
      xc = xc / (1.f + __expf(-xc));                 // SiLU
      sCH[si * SROW + d] = (_Float16)xc;
      sXH[si * SROW + d] = (_Float16)w3;
      w0 = w1; w1 = w2; w2 = w3;
      if (si + 1 < TS) w3 = xb[(size_t)(s0 + si + 1) * D_];
    }
  }
  __syncthreads();

  // Stage 2: wave w = head n. Per (ntile, mtile): 4 WMMAs (i,f use conv path;
  // z,o use raw path), then interleaved float4 store with bias folded in.
  const int lane  = threadIdx.x & 31;
  const int n     = threadIdx.x >> 5;   // head 0..7
  const int lhalf = lane >> 4;
  const int l15   = lane & 15;

  const float* Wi = w_i + (size_t)n * DH_ * DH_;
  const float* Wf = w_f + (size_t)n * DH_ * DH_;
  const float* Wz = w_z + (size_t)n * DH_ * DH_;
  const float* Wo = w_o + (size_t)n * DH_ * DH_;

#pragma unroll
  for (int nt = 0; nt < 2; ++nt) {
    const int o = nt * 16 + l15;
    const v16h Bi = loadB_w(Wi, o, lhalf);
    const v16h Bf = loadB_w(Wf, o, lhalf);
    const v16h Bz = loadB_w(Wz, o, lhalf);
    const v16h Bo = loadB_w(Wo, o, lhalf);
    const float bi = bias[(n * 4 + 0) * DH_ + o];
    const float bf = bias[(n * 4 + 1) * DH_ + o];
    const float bz = bias[(n * 4 + 2) * DH_ + o];
    const float bo = bias[(n * 4 + 3) * DH_ + o];
#pragma unroll
    for (int mt = 0; mt < 2; ++mt) {
      const v16h Ac = loadA_lds(sCH, mt * 16 + l15, lhalf);
      const v16h Ax = loadA_lds(sXH, mt * 16 + l15, lhalf);
      v8f Di = {}, Df = {}, Dz = {}, Do = {};
      Di = wmma_f16(Ac, Bi, Di);
      Df = wmma_f16(Ac, Bf, Df);
      Dz = wmma_f16(Ax, Bz, Dz);
      Do = wmma_f16(Ax, Bo, Do);
      // C/D layout: lane l, vgpr r -> M = r + 8*(l>>4), N = l&15
#pragma unroll
      for (int r = 0; r < 8; ++r) {
        const int sg = s0 + mt * 16 + r + 8 * lhalf;
        float4 v = make_float4(Di[r] + bi, Df[r] + bf, Dz[r] + bz, Do[r] + bo);
        pre4[(((size_t)sg * B_ + b) * NH_ + n) * DH_ + o] = v;
      }
    }
  }
}

// ---------------- Kernel 2: sequential scan + fused GroupNorm ---------------
__global__ __launch_bounds__(32)
void scan_kernel(const float4* __restrict__ pre4,
                 const float* __restrict__ R,
                 const float* __restrict__ gn_w,
                 float* __restrict__ out)
{
  const int wid = blockIdx.x;           // 0..63: one wave per (b, n)
  const int b = wid >> 3;
  const int n = wid & 7;
  const int i = threadIdx.x;            // DH index, one per lane

  // R[n][g][i][0..31] pinned in VGPRs: 128 f32 per lane.
  float Rr[4][DH_];
#pragma unroll
  for (int g = 0; g < 4; ++g) {
    const float4* rp =
        (const float4*)(R + (((size_t)n * 4 + g) * DH_ + i) * DH_);
#pragma unroll
    for (int j4 = 0; j4 < 8; ++j4) {
      const float4 v = rp[j4];
      Rr[g][4 * j4 + 0] = v.x; Rr[g][4 * j4 + 1] = v.y;
      Rr[g][4 * j4 + 2] = v.z; Rr[g][4 * j4 + 3] = v.w;
    }
  }
  const float gw = gn_w[n * DH_ + i];

  const size_t stride = (size_t)B_ * NH_ * DH_;   // float4 elems per step
  const float4* pw = pre4 + ((size_t)b * NH_ + n) * DH_ + i;
  float* outp = out + (size_t)b * S_ * D_ + n * DH_ + i;

  // 8-deep register prefetch ring to hide HBM latency over 4096 serial steps.
  float4 buf[PF];
#pragma unroll
  for (int k = 0; k < PF; ++k) buf[k] = pw[(size_t)k * stride];
  const float4* pnext = pw + (size_t)PF * stride;   // running prefetch ptr

  float h = 0.f, c = 0.f, ns = 0.f, m = -1e30f;

  // One recurrence step + fused GroupNorm; advances outp by one row.
  auto step = [&](const float4 p) {
    // raw = p + R[g][i][:] . h  with split even/odd accumulators (8 chains
    // of depth 16) so the serial FMA chain is not latency-bound at occ=1.
    float a0 = p.x, b0 = p.y, c0 = p.z, d0 = p.w;
    float a1 = 0.f, b1 = 0.f, c1 = 0.f, d1 = 0.f;
#pragma unroll
    for (int j = 0; j < DH_; j += 2) {
      const float hj0 = __uint_as_float(
          __builtin_amdgcn_readlane(__float_as_uint(h), j));
      const float hj1 = __uint_as_float(
          __builtin_amdgcn_readlane(__float_as_uint(h), j + 1));
      a0 = fmaf(Rr[0][j], hj0, a0);  a1 = fmaf(Rr[0][j + 1], hj1, a1);
      b0 = fmaf(Rr[1][j], hj0, b0);  b1 = fmaf(Rr[1][j + 1], hj1, b1);
      c0 = fmaf(Rr[2][j], hj0, c0);  c1 = fmaf(Rr[2][j + 1], hj1, c1);
      d0 = fmaf(Rr[3][j], hj0, d0);  d1 = fmaf(Rr[3][j + 1], hj1, d1);
    }
    const float ri = a0 + a1, rf = b0 + b1, rz = c0 + c1, ro = d0 + d1;

    // mLSTM gating with log-space stabilizer.
    const float lf  = fminf(rf, 0.f) - log1pf(__expf(-fabsf(rf)));
    const float lfm = m + lf;
    const float mn  = fmaxf(ri, lfm);
    const float ig  = __expf(ri - mn);
    const float fg  = __expf(lfm - mn);
    const float cn  = fg * c + ig * tanhf(rz);
    const float nn  = fg * ns + ig;
    const float og  = 1.f / (1.f + __expf(-ro));
    const float hn  = og * cn / nn;
    h = hn; c = cn; ns = nn; m = mn;

    // Fused GroupNorm over DH=32 == the wave.
    float sum = hn;
#pragma unroll
    for (int off = 16; off > 0; off >>= 1) sum += __shfl_xor(sum, off);
    const float mu = sum * (1.f / 32.f);
    const float dv = hn - mu;
    float s2 = dv * dv;
#pragma unroll
    for (int off = 16; off > 0; off >>= 1) s2 += __shfl_xor(s2, off);
    const float inv = rsqrtf(s2 * (1.f / 32.f) + 1e-5f);
    *outp = dv * inv * gw;
    outp += D_;
  };

  // Steady state: no bounds checks, running pointers only.
  for (int s = 0; s < S_ - PF; s += PF) {
#pragma unroll
    for (int u = 0; u < PF; ++u) {
      const float4 p = buf[u];
      buf[u] = *pnext;
      pnext += stride;
      step(p);
    }
  }
  // Epilogue: drain the ring (last PF steps, no prefetch).
#pragma unroll
  for (int u = 0; u < PF; ++u) step(buf[u]);
}

// ---------------------------------------------------------------------------
extern "C" void kernel_launch(void* const* d_in, const int* in_sizes, int n_in,
                              void* d_out, int out_size, void* d_ws, size_t ws_size,
                              hipStream_t stream) {
  const float* x      = (const float*)d_in[0];
  const float* conv_w = (const float*)d_in[1];
  const float* conv_b = (const float*)d_in[2];
  const float* w_i    = (const float*)d_in[3];
  const float* w_f    = (const float*)d_in[4];
  const float* w_z    = (const float*)d_in[5];
  const float* w_o    = (const float*)d_in[6];
  const float* R      = (const float*)d_in[7];
  const float* bias   = (const float*)d_in[8];
  const float* gn_w   = (const float*)d_in[9];
  float*  out  = (float*)d_out;
  float4* pre4 = (float4*)d_ws;   // needs S*B*NH*DH*16 B = 134.2 MB scratch

  (void)in_sizes; (void)n_in; (void)out_size; (void)ws_size;

  precompute_kernel<<<B_ * (S_ / TS), 256, 0, stream>>>(
      x, conv_w, conv_b, w_i, w_f, w_z, w_o, bias, pre4);
  scan_kernel<<<B_ * NH_, 32, 0, stream>>>(pre4, R, gn_w, out);
}
